// MultiHeadAttention_7078106103945
// MI455X (gfx1250) — matrix operand order repaired and checked by reading, not compiled, not run on hardware
//
#include <hip/hip_runtime.h>

// ---------------------------------------------------------------------------
// MHA for B=2, N=2048, DIM=1024, HEADS=16, HEAD_DIM=64 on gfx1250 (wave32).
// Matmuls: v_wmma_f32_16x16x32_bf16 (f32 accumulation).
// Global->LDS staging: global_load_async_to_lds_b128 (ASYNCcnt DMA path),
// double-buffered LDS for load/compute overlap.
// ---------------------------------------------------------------------------

typedef __attribute__((ext_vector_type(16))) __bf16          v16bf;
typedef __attribute__((ext_vector_type(16))) unsigned short  v16u;
typedef __attribute__((ext_vector_type(8)))  unsigned short  v8u;
typedef __attribute__((ext_vector_type(4)))  unsigned short  v4u;
typedef __attribute__((ext_vector_type(8)))  float           v8f;

#define Bsz   2
#define Nseq  2048
#define DIM   1024
#define HEADS 16
#define HD    64

// f32 -> bf16 bits, round-to-nearest-even
static __device__ __forceinline__ unsigned short f2bf(float f) {
  unsigned u = __float_as_uint(f);
  u += 0x7FFFu + ((u >> 16) & 1u);
  return (unsigned short)(u >> 16);
}

static __device__ __forceinline__ v8f vzero8() {
  v8f z;
#pragma unroll
  for (int i = 0; i < 8; ++i) z[i] = 0.0f;
  return z;
}

// Async DMA: copy 32 bytes global -> LDS for this lane (2 x b128).
// INST_OFFSET is applied to BOTH the LDS and global addresses (ISA 08 §4.4),
// so one base pair + offset:16 moves the second chunk.
static __device__ __forceinline__ void async_cp32(void* lds, const void* g) {
  unsigned lo = (unsigned)(unsigned long long)(uintptr_t)lds;  // LDS byte offset
  unsigned long long ga = (unsigned long long)(uintptr_t)g;
  asm volatile(
      "global_load_async_to_lds_b128 %0, %1, off\n\t"
      "global_load_async_to_lds_b128 %0, %1, off offset:16"
      :: "v"(lo), "v"(ga) : "memory");
}
static __device__ __forceinline__ void wait_async_le4() {
  asm volatile("s_wait_asynccnt 0x4" ::: "memory");
}
static __device__ __forceinline__ void wait_async_0() {
  asm volatile("s_wait_asynccnt 0x0" ::: "memory");
}

// A-matrix fragment (16x32 bf16, ISA layout): lane holds row (lane&15);
// lanes 0-15: k = 0..7 and 16..23 ; lanes 16-31: k = 8..15 and 24..31.
static __device__ __forceinline__ v16u fragA(const unsigned short* row, int lane) {
  const int c8 = (lane & 16) ? 8 : 0;
  v8u lo = *(const v8u*)(row + c8);
  v8u hi = *(const v8u*)(row + c8 + 16);
  v16u r;
#pragma unroll
  for (int i = 0; i < 8; ++i) { r[i] = lo[i]; r[i + 8] = hi[i]; }
  return r;
}

// B-matrix fragment (32x16 bf16): lane holds column (lane&15);
// lanes 0-15: k = 0..7 and 16..23 ; lanes 16-31: k = 8..15 and 24..31.
static __device__ __forceinline__ v16u fragB(const unsigned short* col, int lane) {
  const int c8 = (lane & 16) ? 8 : 0;
  v8u lo = *(const v8u*)(col + c8);
  v8u hi = *(const v8u*)(col + c8 + 16);
  v16u r;
#pragma unroll
  for (int i = 0; i < 8; ++i) { r[i] = lo[i]; r[i + 8] = hi[i]; }
  return r;
}

static __device__ __forceinline__ v8f wmma_bf16(v16u a, v16u b, v8f c) {
  return __builtin_amdgcn_wmma_f32_16x16x32_bf16(
      false, __builtin_bit_cast(v16bf, a),
      false, __builtin_bit_cast(v16bf, b),
      (short)0, c, false, false);
}

// ---------------------------------------------------------------------------
// Kernel 1: f32 -> bf16 (4 elements / thread)
// ---------------------------------------------------------------------------
__global__ __launch_bounds__(256) void cvt_bf16(const float* __restrict__ in,
                                                unsigned short* __restrict__ out,
                                                int n4) {
  int i = blockIdx.x * 256 + threadIdx.x;
  if (i < n4) {
    float4 f = ((const float4*)in)[i];
    v4u r;
    r[0] = f2bf(f.x); r[1] = f2bf(f.y); r[2] = f2bf(f.z); r[3] = f2bf(f.w);
    ((v4u*)out)[i] = r;
  }
}

// ---------------------------------------------------------------------------
// Kernel 2: QKV GEMM.  qkv[m,o] = sum_k xb[m,k] * wqkv[o,k]   (y = x @ W^T)
// 128x128 tile per block, 8 waves (4 m-slices x 2 n-slices), k-step 32,
// async double-buffered LDS staging.
// ---------------------------------------------------------------------------
__global__ __launch_bounds__(256) void qkv_gemm(
    const unsigned short* __restrict__ xb,   // [4096,1024] bf16
    const unsigned short* __restrict__ wb,   // [3072,1024] bf16
    unsigned short* __restrict__ Qg,         // [B,H,N,64]
    unsigned short* __restrict__ Kg,         // [B,H,N,64]
    unsigned short* __restrict__ Vtg) {      // [B,H,64,N]
  __shared__ unsigned short ldsA[2][128 * 32];
  __shared__ unsigned short ldsB[2][128 * 32];

  const int o0 = blockIdx.x * 128;
  const int m0 = blockIdx.y * 128;
  const int tid  = threadIdx.x;
  const int lane = tid & 31;
  const int wave = tid >> 5;
  const int wm = (wave & 3) * 32;   // wave's m offset in tile
  const int wn = (wave >> 2) * 64;  // wave's n offset in tile
  const int ln15 = lane & 15;
  const int hi8  = (lane & 16) ? 8 : 0;

  v8f acc[2][4];
#pragma unroll
  for (int i = 0; i < 2; ++i)
#pragma unroll
    for (int j = 0; j < 4; ++j) acc[i][j] = vzero8();

  const int lr = tid >> 1;        // 0..127 : row for cooperative staging
  const int lh = (tid & 1) * 16;  // 0 or 16 : k-half

  const unsigned short* gA = &xb[(m0 + lr) * DIM + lh];
  const unsigned short* gB = &wb[(o0 + lr) * DIM + lh];
  const int soff = lr * 32 + lh;

  // prologue: stage k-block 0 into buffer 0 (4 async ops / lane)
  async_cp32(&ldsA[0][soff], gA);
  async_cp32(&ldsB[0][soff], gB);

  for (int it = 0; it < DIM / 32; ++it) {
    const int cur = it & 1;
    __syncthreads();  // everyone done reading buf[cur^1] from it-1
    if (it + 1 < DIM / 32) {
      async_cp32(&ldsA[cur ^ 1][soff], gA + (it + 1) * 32);
      async_cp32(&ldsB[cur ^ 1][soff], gB + (it + 1) * 32);
      wait_async_le4();  // drain buf[cur]'s 4 ops; next 4 stay in flight
    } else {
      wait_async_0();
    }
    __syncthreads();  // buf[cur] globally visible

    v16u a0 = fragA(&ldsA[cur][(wm + ln15) * 32], lane);
    v16u a1 = fragA(&ldsA[cur][(wm + 16 + ln15) * 32], lane);
#pragma unroll
    for (int j = 0; j < 4; ++j) {
      v16u b = fragB(&ldsB[cur][(wn + j * 16 + ln15) * 32], lane);
      acc[0][j] = wmma_bf16(a0, b, acc[0][j]);
      acc[1][j] = wmma_bf16(a1, b, acc[1][j]);
    }
  }

#pragma unroll
  for (int i = 0; i < 2; ++i)
#pragma unroll
    for (int j = 0; j < 4; ++j)
#pragma unroll
      for (int v = 0; v < 8; ++v) {
        int m = m0 + wm + i * 16 + v + hi8;
        int o = o0 + wn + j * 16 + ln15;
        unsigned short bv = f2bf(acc[i][j][v]);
        int b = m >> 11, n = m & (Nseq - 1);
        if (o < DIM) {
          int h = o >> 6, d = o & 63;
          Qg[((b * HEADS + h) * Nseq + n) * HD + d] = bv;
        } else if (o < 2 * DIM) {
          int o2 = o - DIM; int h = o2 >> 6, d = o2 & 63;
          Kg[((b * HEADS + h) * Nseq + n) * HD + d] = bv;
        } else {
          int o2 = o - 2 * DIM; int h = o2 >> 6, d = o2 & 63;
          Vtg[((b * HEADS + h) * HD + d) * Nseq + n] = bv;  // transposed V
        }
      }
}

// ---------------------------------------------------------------------------
// Kernel 3: flash attention.  Block = one (b,h), 128 queries; 8 waves x 16 q.
// Online softmax over key blocks of 64; async double-buffered K/V^T staging.
// scale = 1/sqrt(64) = 0.125.
// ---------------------------------------------------------------------------
__global__ __launch_bounds__(256) void flash_attn(
    const unsigned short* __restrict__ Qg,   // [B*H, N, 64]
    const unsigned short* __restrict__ Kg,   // [B*H, N, 64]
    const unsigned short* __restrict__ Vtg,  // [B*H, 64, N]
    unsigned short* __restrict__ outb) {     // [B, N, 1024] bf16
  __shared__ unsigned short ldsK[2][64 * 64];    // keys  [key][d]
  __shared__ unsigned short ldsV[2][64 * 64];    // V^T   [d][key]
  __shared__ unsigned short ldsP[8][16 * 64];    // per-wave P tile [16][64]

  const int bh   = blockIdx.x;   // 0..31
  const int qb   = blockIdx.y;   // 0..15
  const int tid  = threadIdx.x;
  const int lane = tid & 31;
  const int wave = tid >> 5;
  const int q0   = qb * 128 + wave * 16;
  const int ln15 = lane & 15;
  const int hi8  = (lane & 16) ? 8 : 0;

  // Q fragments for this wave's 16 query rows (d = 0..31 and 32..63)
  const unsigned short* qrow = Qg + ((size_t)bh * Nseq + q0 + ln15) * HD;
  v16u qa0 = fragA(qrow, lane);
  v16u qa1 = fragA(qrow + 32, lane);

  v8f o[4];
  float mrow[8], lrow[8];
#pragma unroll
  for (int t = 0; t < 4; ++t) o[t] = vzero8();
#pragma unroll
  for (int v = 0; v < 8; ++v) { mrow[v] = -3.0e38f; lrow[v] = 0.0f; }

  const int cr = tid >> 2;         // 0..63 : row for cooperative staging
  const int cq = (tid & 3) * 16;   // 0,16,32,48
  const unsigned short* gK = &Kg[((size_t)bh * Nseq + cr) * HD + cq];
  const unsigned short* gV = &Vtg[((size_t)bh * HD + cr) * Nseq + cq];
  const int soff = cr * 64 + cq;

  // prologue: stage key-block 0 (4 async ops / lane)
  async_cp32(&ldsK[0][soff], gK);
  async_cp32(&ldsV[0][soff], gV);

  for (int it = 0; it < Nseq / 64; ++it) {
    const int cur = it & 1;
    __syncthreads();
    if (it + 1 < Nseq / 64) {
      async_cp32(&ldsK[cur ^ 1][soff], gK + (size_t)(it + 1) * 64 * HD);
      async_cp32(&ldsV[cur ^ 1][soff], gV + (it + 1) * 64);
      wait_async_le4();
    } else {
      wait_async_0();
    }
    __syncthreads();

    // S = Q K^T (4 tiles of 16 keys), f32 accum, scaled
    v8f s[4];
#pragma unroll
    for (int j = 0; j < 4; ++j) {
      v8f z = vzero8();
      z = wmma_bf16(qa0, fragB(&ldsK[cur][(j * 16 + ln15) * 64], lane), z);
      z = wmma_bf16(qa1, fragB(&ldsK[cur][(j * 16 + ln15) * 64 + 32], lane), z);
      s[j] = z * 0.125f;
    }

    // Online softmax. C layout: vgpr v -> row (v + hi8), lane&15 -> col.
#pragma unroll
    for (int v = 0; v < 8; ++v) {
      float t = fmaxf(fmaxf(s[0][v], s[1][v]), fmaxf(s[2][v], s[3][v]));
#pragma unroll
      for (int off = 1; off < 16; off <<= 1)
        t = fmaxf(t, __shfl_xor(t, off, 32));
      float mN    = fmaxf(mrow[v], t);
      float alpha = __expf(mrow[v] - mN);
      mrow[v] = mN;
      float rs = 0.0f;
#pragma unroll
      for (int j = 0; j < 4; ++j) {
        float p = __expf(s[j][v] - mN);
        s[j][v] = p;
        rs += p;
      }
#pragma unroll
      for (int off = 1; off < 16; off <<= 1)
        rs += __shfl_xor(rs, off, 32);
      lrow[v] = lrow[v] * alpha + rs;
#pragma unroll
      for (int t4 = 0; t4 < 4; ++t4) o[t4][v] *= alpha;
    }

    // P (C layout) -> row-major bf16 tile in wave-private LDS
    unsigned short* P = &ldsP[wave][0];
#pragma unroll
    for (int j = 0; j < 4; ++j)
#pragma unroll
      for (int v = 0; v < 8; ++v)
        P[(v + hi8) * 64 + j * 16 + ln15] = f2bf(s[j][v]);
    // cross-lane LDS RAW within the wave: drain DS counter (CDNA5 split wait)
    asm volatile("s_wait_dscnt 0x0" ::: "memory");

    v16u pa0 = fragA(&P[ln15 * 64], lane);        // keys 0..31
    v16u pa1 = fragA(&P[ln15 * 64 + 32], lane);   // keys 32..63
#pragma unroll
    for (int t4 = 0; t4 < 4; ++t4) {
      o[t4] = wmma_bf16(pa0, fragB(&ldsV[cur][(t4 * 16 + ln15) * 64], lane), o[t4]);
      o[t4] = wmma_bf16(pa1, fragB(&ldsV[cur][(t4 * 16 + ln15) * 64 + 32], lane), o[t4]);
    }
  }

  // normalize and store bf16 at [b, n, h*64 + d]
  const int b = bh >> 4, h = bh & 15;
#pragma unroll
  for (int t4 = 0; t4 < 4; ++t4)
#pragma unroll
    for (int v = 0; v < 8; ++v) {
      int n   = q0 + v + hi8;
      int col = h * HD + t4 * 16 + ln15;
      float val = o[t4][v] / lrow[v];
      outb[((size_t)b * Nseq + n) * DIM + col] = f2bf(val);
    }
}

// ---------------------------------------------------------------------------
// Kernel 4: projection GEMM + bias.  out[m,o] = sum_k a[m,k]*wproj[o,k] + b[o]
// ---------------------------------------------------------------------------
__global__ __launch_bounds__(256) void proj_gemm(
    const unsigned short* __restrict__ ab,   // [4096,1024] bf16
    const unsigned short* __restrict__ wb,   // [1024,1024] bf16
    const float* __restrict__ bias,          // [1024]
    float* __restrict__ out) {               // [4096,1024] f32
  __shared__ unsigned short ldsA[2][128 * 32];
  __shared__ unsigned short ldsB[2][128 * 32];

  const int o0 = blockIdx.x * 128;
  const int m0 = blockIdx.y * 128;
  const int tid  = threadIdx.x;
  const int lane = tid & 31;
  const int wave = tid >> 5;
  const int wm = (wave & 3) * 32;
  const int wn = (wave >> 2) * 64;
  const int ln15 = lane & 15;
  const int hi8  = (lane & 16) ? 8 : 0;

  v8f acc[2][4];
#pragma unroll
  for (int i = 0; i < 2; ++i)
#pragma unroll
    for (int j = 0; j < 4; ++j) acc[i][j] = vzero8();

  const int lr = tid >> 1;
  const int lh = (tid & 1) * 16;
  const unsigned short* gA = &ab[(m0 + lr) * DIM + lh];
  const unsigned short* gB = &wb[(o0 + lr) * DIM + lh];
  const int soff = lr * 32 + lh;

  async_cp32(&ldsA[0][soff], gA);
  async_cp32(&ldsB[0][soff], gB);

  for (int it = 0; it < DIM / 32; ++it) {
    const int cur = it & 1;
    __syncthreads();
    if (it + 1 < DIM / 32) {
      async_cp32(&ldsA[cur ^ 1][soff], gA + (it + 1) * 32);
      async_cp32(&ldsB[cur ^ 1][soff], gB + (it + 1) * 32);
      wait_async_le4();
    } else {
      wait_async_0();
    }
    __syncthreads();

    v16u a0 = fragA(&ldsA[cur][(wm + ln15) * 32], lane);
    v16u a1 = fragA(&ldsA[cur][(wm + 16 + ln15) * 32], lane);
#pragma unroll
    for (int j = 0; j < 4; ++j) {
      v16u b = fragB(&ldsB[cur][(wn + j * 16 + ln15) * 32], lane);
      acc[0][j] = wmma_bf16(a0, b, acc[0][j]);
      acc[1][j] = wmma_bf16(a1, b, acc[1][j]);
    }
  }

#pragma unroll
  for (int i = 0; i < 2; ++i)
#pragma unroll
    for (int j = 0; j < 4; ++j) {
      int oc = o0 + wn + j * 16 + ln15;
      float bb = bias[oc];
#pragma unroll
      for (int v = 0; v < 8; ++v) {
        int m = m0 + wm + i * 16 + v + hi8;
        out[(size_t)m * DIM + oc] = acc[i][j][v] + bb;
      }
    }
}

// ---------------------------------------------------------------------------
// Host-side launcher
// ---------------------------------------------------------------------------
extern "C" void kernel_launch(void* const* d_in, const int* in_sizes, int n_in,
                              void* d_out, int out_size, void* d_ws, size_t ws_size,
                              hipStream_t stream) {
  const float* x      = (const float*)d_in[0];  // [2,2048,1024]
  const float* w_qkv  = (const float*)d_in[1];  // [3072,1024]
  const float* w_proj = (const float*)d_in[2];  // [1024,1024]
  const float* b_proj = (const float*)d_in[3];  // [1024]
  float* out = (float*)d_out;                   // [2,2048,1024]

  char* ws = (char*)d_ws;
  unsigned short* xb     = (unsigned short*)(ws);              //  8 MB
  unsigned short* wqkvb  = (unsigned short*)(ws +  8388608);   //  6 MB
  unsigned short* wprojb = (unsigned short*)(ws + 14680064);   //  2 MB
  unsigned short* Qg     = (unsigned short*)(ws + 16777216);   //  8 MB
  unsigned short* Kg     = (unsigned short*)(ws + 25165824);   //  8 MB
  unsigned short* Vtg    = (unsigned short*)(ws + 33554432);   //  8 MB
  unsigned short* attnb  = (unsigned short*)(ws + 41943040);   //  8 MB

  // 1) f32 -> bf16 staging
  cvt_bf16<<<(4194304 / 4) / 256, 256, 0, stream>>>(x, xb, 4194304 / 4);
  cvt_bf16<<<(3145728 / 4) / 256, 256, 0, stream>>>(w_qkv, wqkvb, 3145728 / 4);
  cvt_bf16<<<(1048576 / 4) / 256, 256, 0, stream>>>(w_proj, wprojb, 1048576 / 4);

  // 2) QKV GEMM + scatter (V stored transposed)
  qkv_gemm<<<dim3(3 * DIM / 128, (Bsz * Nseq) / 128), 256, 0, stream>>>(
      xb, wqkvb, Qg, Kg, Vtg);

  // 3) flash attention
  flash_attn<<<dim3(Bsz * HEADS, Nseq / 128), 256, 0, stream>>>(Qg, Kg, Vtg, attnb);

  // 4) projection + bias
  proj_gemm<<<dim3(DIM / 128, (Bsz * Nseq) / 128), 256, 0, stream>>>(
      attnb, wprojb, b_proj, out);
}
